// CategoricalLogLikelihoodLossLayer_15101105013077
// MI455X (gfx1250) — matrix operand order repaired and checked by reading, not compiled, not run on hardware
//
#include <hip/hip_runtime.h>
#include <hip/hip_bf16.h>

// CDNA5 / gfx1250, wave32. v_wmma_f32_16x16x32_bf16 with bf16 hi/lo split
// (3 products, f32 accumulate) -> ~fp32-quality Mahalanobis GEMM at bf16 rate.
// Round 2: 32 rows/wave (2 A-tiles) + 4 independent accumulator chains to
// break the WMMA RAW chain and halve LDS-staging overhead.

typedef __attribute__((ext_vector_type(16))) __bf16        v16bf;
typedef __attribute__((ext_vector_type(8)))  float         v8f;
typedef __attribute__((ext_vector_type(16))) unsigned short v16us;
typedef __attribute__((ext_vector_type(8)))  unsigned int   v8u;

#define DIM 128
#define KSTATES 16

__device__ __forceinline__ void bf16_split(float v, unsigned short& hi, unsigned short& lo) {
    unsigned u  = __float_as_uint(v);
    hi          = (unsigned short)(u >> 16);                 // truncate-to-bf16
    float fh    = __uint_as_float(u & 0xffff0000u);
    unsigned ul = __float_as_uint(v - fh);                   // exact residual
    lo          = (unsigned short)(ul >> 16);
}

// ---------------------------------------------------------------------------
// Kernel A: per-state Cholesky -> W = L^{-1} (bf16 hi/lo, stored as W^T),
//           c = W mu, beta = 0.5*||c||^2 + sum(log diag L).  16 blocks.
// ---------------------------------------------------------------------------
__global__ __launch_bounds__(128)
void setup_kernel(const float* __restrict__ mu, const float* __restrict__ sigma,
                  unsigned short* __restrict__ wt_hi, unsigned short* __restrict__ wt_lo,
                  float* __restrict__ cvec, float* __restrict__ beta) {
    __shared__ float S[DIM * DIM];                            // 64 KB
    const int k = blockIdx.x;
    const int tid = threadIdx.x;

    const float* sg = sigma + (size_t)k * DIM * DIM;
    for (int j = 0; j < DIM; ++j) S[j * DIM + tid] = sg[j * DIM + tid];
    __syncthreads();

    // Left-looking Cholesky; dot products computed in parallel across rows.
    for (int j = 0; j < DIM; ++j) {
        float dot = 0.f;
        if (tid >= j) {
            for (int p = 0; p < j; ++p) dot += S[tid * DIM + p] * S[j * DIM + p];
        }
        if (tid == j) S[j * DIM + j] = sqrtf(S[j * DIM + j] - dot);
        __syncthreads();
        if (tid > j) S[tid * DIM + j] = (S[tid * DIM + j] - dot) / S[j * DIM + j];
        __syncthreads();
    }

    const float ldterm = logf(S[tid * DIM + tid]);            // log diag(L), one per thread

    // Thread c computes column c of W = L^{-1} (forward substitution).
    const int c = tid;
    float w[DIM];                                             // private (scratch) column
    w[c] = 1.0f / S[c * DIM + c];
    for (int r = c + 1; r < DIM; ++r) {
        float s = 0.f;
        for (int p = c; p < r; ++p) s += S[r * DIM + p] * w[p];
        w[r] = -s / S[r * DIM + r];
    }

    // W^T row c (= kd index c, over n): contiguous, pre-split to bf16 hi/lo.
    {
        const size_t base = ((size_t)k * DIM + c) * DIM;
        for (int r = 0; r < DIM; ++r) {
            float v = (r < c) ? 0.f : w[r];
            unsigned short hs, ls;
            bf16_split(v, hs, ls);
            wt_hi[base + r] = hs;
            wt_lo[base + r] = ls;
        }
    }
    __syncthreads();                                          // everyone done reading L
    for (int r = 0; r < DIM; ++r) S[c * DIM + r] = (r < c) ? 0.f : w[r];  // S := W^T
    __syncthreads();

    // cvec[n] = sum_d W^T[d][n] * mu[d]   (thread n; lane-consecutive LDS reads)
    const float* mug = mu + (size_t)k * DIM;
    float cv = 0.f;
    for (int d = 0; d < DIM; ++d) cv += S[d * DIM + tid] * mug[d];
    cvec[(size_t)k * DIM + tid] = cv;

    float val = 0.5f * cv * cv + ldterm;
    __syncthreads();                                          // S free for reduction scratch
    val += __shfl_xor(val, 1, 32);
    val += __shfl_xor(val, 2, 32);
    val += __shfl_xor(val, 4, 32);
    val += __shfl_xor(val, 8, 32);
    val += __shfl_xor(val, 16, 32);
    if ((tid & 31) == 0) S[tid >> 5] = val;
    __syncthreads();
    if (tid == 0) beta[k] = S[0] + S[1] + S[2] + S[3];
}

// ---------------------------------------------------------------------------
// Kernel B: per wave = 32 rows of X (2 A-tiles). Loop k: stage W_k^T (hi+lo
// bf16) into LDS pre-swizzled as B fragments; 8 col-tiles x 4 k-steps x
// (2 tiles x 3 products) WMMAs over 4 independent accumulator chains.
// Fold 0.5*y^2 - c.y, weight by probs. One partial per block.
// ---------------------------------------------------------------------------
__global__ __launch_bounds__(256)
void gmm_wmma_kernel(const float* __restrict__ x, const float* __restrict__ probs,
                     const unsigned short* __restrict__ wt_hi,
                     const unsigned short* __restrict__ wt_lo,
                     const float* __restrict__ cvec, const float* __restrict__ beta,
                     float* __restrict__ partials) {
    __shared__ unsigned int smem[16384];                      // 64 KB: hi [0,8192) dw, lo [8192,16384)
    const int tid    = threadIdx.x;
    const int wave   = tid >> 5;
    const int lane   = tid & 31;
    const int half16 = lane >> 4;                             // 0 or 1
    const int mrow   = lane & 15;
    const long rowbase = (long)blockIdx.x * 256 + (long)wave * 32;

    // A fragments: 16x32 bf16 per k-step, ISA layout:
    //   halves 0..7  -> K = s*32 +      half16*8 + e
    //   halves 8..15 -> K = s*32 + 16 + half16*8 + e
    v16bf ahi[2][4], alo[2][4];
    #pragma unroll
    for (int t = 0; t < 2; ++t) {
        const float4* rp4 = (const float4*)(x + (rowbase + t * 16 + mrow) * (long)DIM);
        #pragma unroll
        for (int s = 0; s < 4; ++s) {
            float f0[8], f1[8];
            *(float4*)&f0[0] = rp4[s * 8 + half16 * 2];       // K = s*32 + half16*8 ..
            *(float4*)&f0[4] = rp4[s * 8 + half16 * 2 + 1];
            *(float4*)&f1[0] = rp4[s * 8 + half16 * 2 + 4];   // K = s*32 + 16 + half16*8 ..
            *(float4*)&f1[4] = rp4[s * 8 + half16 * 2 + 5];
            v16us uh, ul;
            #pragma unroll
            for (int e = 0; e < 8; ++e) {
                unsigned short h, l;
                bf16_split(f0[e], h, l); uh[e]     = h; ul[e]     = l;
                bf16_split(f1[e], h, l); uh[8 + e] = h; ul[8 + e] = l;
            }
            ahi[t][s] = __builtin_bit_cast(v16bf, uh);
            alo[t][s] = __builtin_bit_cast(v16bf, ul);
        }
    }

    float wavesum = 0.f;
    for (int k = 0; k < KSTATES; ++k) {
        __syncthreads();                                      // previous k's LDS reads done
        // Stage W_k^T pre-swizzled: fragment block (mat,s,n,lane) = 32 bytes.
        // B layout (dense 32x16 bf16): lane&15 -> N, element e -> K = 8*(lane>>4) + (e&7) + 16*(e>>3).
        #pragma unroll
        for (int i = 0; i < 8; ++i) {
            const int fb  = i * 256 + tid;                    // 0..2047
            const int mat = fb >> 10;
            const int r   = fb & 1023;
            const int s   = r >> 8;
            const int n   = (r >> 5) & 7;
            const int ln  = r & 31;
            const int kd  = s * 32 + ((ln >> 4) << 3);
            const unsigned short* src =
                (mat ? wt_lo : wt_hi) + (((size_t)k * DIM + kd) * DIM + n * 16 + (ln & 15));
            v16us tmp;
            #pragma unroll
            for (int e = 0; e < 16; ++e) tmp[e] = src[(size_t)((e & 7) + ((e >> 3) << 4)) * DIM];
            *(v8u*)&smem[(size_t)fb * 8] = __builtin_bit_cast(v8u, tmp);
        }
        __syncthreads();

        const float bk = beta[k];
        float rowacc[2][8] = {{0.f,0.f,0.f,0.f,0.f,0.f,0.f,0.f},
                              {0.f,0.f,0.f,0.f,0.f,0.f,0.f,0.f}};

        #pragma unroll
        for (int n = 0; n < 8; ++n) {
            v8f c0t0 = {0.f,0.f,0.f,0.f,0.f,0.f,0.f,0.f};     // tile0: hi*hi
            v8f c0t1 = {0.f,0.f,0.f,0.f,0.f,0.f,0.f,0.f};     // tile1: hi*hi
            v8f c1t0 = {0.f,0.f,0.f,0.f,0.f,0.f,0.f,0.f};     // tile0: cross terms
            v8f c1t1 = {0.f,0.f,0.f,0.f,0.f,0.f,0.f,0.f};     // tile1: cross terms
            #pragma unroll
            for (int s = 0; s < 4; ++s) {
                const int fo = ((s * 8 + n) * 32 + lane) * 8;
                v16bf bhi = __builtin_bit_cast(v16bf, *(const v8u*)&smem[fo]);
                v16bf blo = __builtin_bit_cast(v16bf, *(const v8u*)&smem[fo + 8192]);
                // 6 WMMAs over 4 chains: RAW distance >= 2 everywhere.
                c0t0 = __builtin_amdgcn_wmma_f32_16x16x32_bf16(false, ahi[0][s], false, bhi,
                                                               (short)0, c0t0, false, false);
                c0t1 = __builtin_amdgcn_wmma_f32_16x16x32_bf16(false, ahi[1][s], false, bhi,
                                                               (short)0, c0t1, false, false);
                c1t0 = __builtin_amdgcn_wmma_f32_16x16x32_bf16(false, ahi[0][s], false, blo,
                                                               (short)0, c1t0, false, false);
                c1t1 = __builtin_amdgcn_wmma_f32_16x16x32_bf16(false, ahi[1][s], false, blo,
                                                               (short)0, c1t1, false, false);
                c1t0 = __builtin_amdgcn_wmma_f32_16x16x32_bf16(false, alo[0][s], false, bhi,
                                                               (short)0, c1t0, false, false);
                c1t1 = __builtin_amdgcn_wmma_f32_16x16x32_bf16(false, alo[1][s], false, bhi,
                                                               (short)0, c1t1, false, false);
            }
            const float cj = cvec[k * DIM + n * 16 + mrow];   // column j = n*16 + (lane&15)
            #pragma unroll
            for (int v = 0; v < 8; ++v) {
                const float y0 = c0t0[v] + c1t0[v];
                const float y1 = c0t1[v] + c1t1[v];
                rowacc[0][v] += (0.5f * y0 - cj) * y0;        // 0.5*y^2 - c_j*y
                rowacc[1][v] += (0.5f * y1 - cj) * y1;
            }
        }

        // Reduce each C register across its 16-lane half (cols), weight by p.
        #pragma unroll
        for (int t = 0; t < 2; ++t) {
            #pragma unroll
            for (int v = 0; v < 8; ++v) {
                float m = rowacc[t][v];
                m += __shfl_xor(m, 1, 32);
                m += __shfl_xor(m, 2, 32);
                m += __shfl_xor(m, 4, 32);
                m += __shfl_xor(m, 8, 32);                    // stays within 16-lane halves
                const int row = v + (half16 << 3);            // C: M = v (+8 for lanes 16-31)
                const float p = probs[(rowbase + t * 16 + row) * KSTATES + k];
                wavesum += p * (m + bk);                      // replicated 16x per row
            }
        }
    }

    // Full-wave reduce; each row term was counted 16x.
    wavesum += __shfl_xor(wavesum, 1, 32);
    wavesum += __shfl_xor(wavesum, 2, 32);
    wavesum += __shfl_xor(wavesum, 4, 32);
    wavesum += __shfl_xor(wavesum, 8, 32);
    wavesum += __shfl_xor(wavesum, 16, 32);
    __syncthreads();
    float* red = (float*)smem;
    if (lane == 0) red[wave] = wavesum * (1.0f / 16.0f);
    __syncthreads();
    if (tid == 0) {
        float t = 0.f;
        for (int i = 0; i < 8; ++i) t += red[i];
        partials[blockIdx.x] = t;
    }
}

// ---------------------------------------------------------------------------
// Kernel C: deterministic fixed-order reduction of per-block partials.
// loss = T*c0 + (1/B) * sum partials,   c0 = 0.5*D*ln(2*pi)
// ---------------------------------------------------------------------------
__global__ __launch_bounds__(256)
void reduce_kernel(const float* __restrict__ partials, int n, float* __restrict__ out,
                   float addend, float invB) {
    __shared__ float red[256];
    const int tid = threadIdx.x;
    float s = 0.f;
    for (int i = tid; i < n; i += 256) s += partials[i];
    red[tid] = s;
    __syncthreads();
    for (int off = 128; off > 0; off >>= 1) {
        if (tid < off) red[tid] += red[tid + off];
        __syncthreads();
    }
    if (tid == 0) out[0] = addend + red[0] * invB;
}

extern "C" void kernel_launch(void* const* d_in, const int* in_sizes, int n_in,
                              void* d_out, int out_size, void* d_ws, size_t ws_size,
                              hipStream_t stream) {
    (void)n_in; (void)out_size; (void)ws_size;
    const float* x     = (const float*)d_in[0];   // [B,T,D]
    const float* mu    = (const float*)d_in[1];   // [K,D]
    const float* sigma = (const float*)d_in[2];   // [K,D,D]
    const float* probs = (const float*)d_in[3];   // [B,T,K]
    float* out = (float*)d_out;

    const int BT = in_sizes[0] / DIM;             // 131072
    const int Bb = 64;                            // batch (per reference)
    const int T  = BT / Bb;                       // 2048

    // Workspace layout (~1.06 MB):
    unsigned char* ws = (unsigned char*)d_ws;
    unsigned short* wt_hi = (unsigned short*)(ws);                    // 512 KB
    unsigned short* wt_lo = (unsigned short*)(ws + 524288);           // 512 KB
    float* cvec     = (float*)(ws + 1048576);                         // K*D f32
    float* beta     = (float*)(ws + 1048576 + 8192);                  // K f32
    float* partials = (float*)(ws + 1048576 + 8192 + 256);            // nblocks f32

    const int nblocks = BT / 256;                 // 512, 8 waves x 32 rows each

    setup_kernel<<<KSTATES, 128, 0, stream>>>(mu, sigma, wt_hi, wt_lo, cvec, beta);
    gmm_wmma_kernel<<<nblocks, 256, 0, stream>>>(x, probs, wt_hi, wt_lo, cvec, beta, partials);

    const double c0 = 0.5 * (double)DIM * 1.8378770664093453;  // 0.5*D*ln(2*pi)
    reduce_kernel<<<1, 256, 0, stream>>>(partials, nblocks, out,
                                         (float)((double)T * c0), 1.0f / (float)Bb);
}
